// Sequence_8847632629912
// MI455X (gfx1250) — matrix operand order repaired and checked by reading, not compiled, not run on hardware
//
#include <hip/hip_runtime.h>

// ---- types ---------------------------------------------------------------
typedef __attribute__((ext_vector_type(16))) _Float16 v16h;
typedef __attribute__((ext_vector_type(8)))  _Float16 v8h;
typedef __attribute__((ext_vector_type(8)))  float    v8f;

#define HDIM 64          // hidden size
#define BT   16          // batch-tile rows per workgroup
#define NW   8           // waves per workgroup (wave32)
#define SGS  260         // padded gate-row stride (floats): 8-row bank shift = 32
#define HS   72          // padded h-row stride (halves): breaks bank conflicts

// ---- helpers -------------------------------------------------------------
__device__ __forceinline__ v8f wmma_f16(v16h a, v16h b, v8f c) {
  // 8 args: (neg_a, A, neg_b, B, c_mod, C, reuse_a, reuse_b)
  return __builtin_amdgcn_wmma_f32_16x16x32_f16(false, a, false, b,
                                                (short)0, c, false, false);
}

__device__ __forceinline__ float fast_sigmoid(float x) {
  return __builtin_amdgcn_rcpf(1.0f + __expf(-x));
}
__device__ __forceinline__ float fast_tanh(float x) {
  float e = __expf(2.0f * x);
  return 1.0f - 2.0f * __builtin_amdgcn_rcpf(e + 1.0f);
}

// A-fragment (16x32 f16) in this code's contiguous k order (B matches):
//   lanes 0-15 : row M=lane,    e0..15 -> K=koff+0..15
//   lanes 16-31: row M=lane-16, e0..15 -> K=koff+16..31
__device__ __forceinline__ v16h load_afrag(const _Float16* sh, int kf) {
  const int ln   = threadIdx.x & 31;
  const int row  = ln & 15;
  const int half = ln >> 4;
  const _Float16* p = sh + row * HS + kf * 32 + half * 16;
  v8h lo = *(const v8h*)(p);
  v8h hi = *(const v8h*)(p + 8);
  v16h a;
#pragma unroll
  for (int i = 0; i < 8; ++i) { a[i] = lo[i]; a[i + 8] = hi[i]; }
  return a;
}

// C/D tile (16x16 f32): lanes 0-15 N=lane, M=vgpr; lanes 16-31 N=lane-16, M=vgpr+8
__device__ __forceinline__ void store_ctile(float* sg, v8f c, int ncol0) {
  const int ln = threadIdx.x & 31;
  const int n  = ncol0 + (ln & 15);
  const int mb = (ln >> 4) * 8;
#pragma unroll
  for (int v = 0; v < 8; ++v) sg[(mb + v) * SGS + n] = c[v];
}

// ---- kernel --------------------------------------------------------------
__global__ void __launch_bounds__(NW * 32, 1)
lstm2_wmma_kernel(const float* __restrict__ input,  // (B,T)
                  const float* __restrict__ Wih1,   // (256,1)
                  const float* __restrict__ Whh1,   // (256,64)
                  const float* __restrict__ b1,     // (256,)
                  const float* __restrict__ Wih2,   // (256,64)
                  const float* __restrict__ Whh2,   // (256,64)
                  const float* __restrict__ b2,     // (256,)
                  const float* __restrict__ Wout,   // (1,64)
                  const float* __restrict__ bout,   // (1,)
                  float* __restrict__ out,          // (B,T)
                  int T) {
  __shared__ __align__(16) _Float16 sh1[BT * HS];
  __shared__ __align__(16) _Float16 sh2[BT * HS];
  __shared__ __align__(16) float    sg[BT * SGS];
  __shared__ float sb1[256], swih1[256], sb2[256], swout[64];

  const int tid = threadIdx.x;          // 0..255
  const int wv  = tid >> 5;             // wave id 0..7
  const int ln  = tid & 31;
  const int gate    = wv >> 1;          // i,f,g,o gate this wave owns
  const int ncolbase = gate * 64 + (wv & 1) * 32;  // 32-col slice within 256

  const int bbase = blockIdx.x * BT;

  // stage small constants into LDS, zero h state
  for (int i = tid; i < 256; i += NW * 32) {
    sb1[i] = b1[i]; swih1[i] = Wih1[i]; sb2[i] = b2[i];
  }
  if (tid < 64) swout[tid] = Wout[tid];
  for (int i = tid; i < BT * HS; i += NW * 32) {
    sh1[i] = (_Float16)0.0f; sh2[i] = (_Float16)0.0f;
  }

  // ---- pin weight B-fragments (32x16 f16, K x N) in registers ----
  // lane L<16: col N=L,  e -> K=kb+e ; lane L>=16: col N=L-16, e -> K=kb+16+e
  const int nlane = ln & 15;
  const int khalf = (ln >> 4) * 16;
  v16h Bhh1[2][2], Bih2[2][2], Bhh2[2][2];  // [ntile][kfrag]
#pragma unroll
  for (int nt = 0; nt < 2; ++nt) {
    const int n = ncolbase + nt * 16 + nlane;   // gate column (row of W)
#pragma unroll
    for (int kf = 0; kf < 2; ++kf) {
      const int kb = kf * 32 + khalf;
#pragma unroll
      for (int e = 0; e < 16; ++e) {
        Bhh1[nt][kf][e] = (_Float16)Whh1[n * HDIM + kb + e];
        Bih2[nt][kf][e] = (_Float16)Wih2[n * HDIM + kb + e];
        Bhh2[nt][kf][e] = (_Float16)Whh2[n * HDIM + kb + e];
      }
    }
  }

  // elementwise mapping: 256 lanes over a 16x64 tile, 4 cols/lane
  const int er = tid >> 4;              // row 0..15
  const int ec = (tid & 15) * 4;        // col base
  float c1[4] = {0, 0, 0, 0}, c2[4] = {0, 0, 0, 0};
  const float bo = bout[0];

  __syncthreads();

  for (int t = 0; t < T; ++t) {
    // ================= layer 1 matmul: sg = h1 @ Whh1^T =================
    {
      v16h a0 = load_afrag(sh1, 0);
      v16h a1 = load_afrag(sh1, 1);
#pragma unroll
      for (int nt = 0; nt < 2; ++nt) {
        v8f c = 0.0f;
        c = wmma_f16(a0, Bhh1[nt][0], c);
        c = wmma_f16(a1, Bhh1[nt][1], c);
        store_ctile(sg, c, ncolbase + nt * 16);
      }
    }
    __syncthreads();

    // ================= layer 1 elementwise ==============================
    {
      const float x = input[(size_t)(bbase + er) * T + t];
      const float* gr = sg + er * SGS;
#pragma unroll
      for (int j = 0; j < 4; ++j) {
        const int c0 = ec + j;
        float ig = gr[c0]       + sb1[c0]       + x * swih1[c0];
        float fg = gr[64 + c0]  + sb1[64 + c0]  + x * swih1[64 + c0];
        float gg = gr[128 + c0] + sb1[128 + c0] + x * swih1[128 + c0];
        float og = gr[192 + c0] + sb1[192 + c0] + x * swih1[192 + c0];
        float cn = fast_sigmoid(fg) * c1[j] + fast_sigmoid(ig) * fast_tanh(gg);
        c1[j] = cn;
        sh1[er * HS + c0] = (_Float16)(fast_sigmoid(og) * fast_tanh(cn));
      }
    }
    __syncthreads();

    // ================= layer 2 matmul: sg = h1@Wih2^T + h2@Whh2^T =======
    {
      v16h a10 = load_afrag(sh1, 0);
      v16h a11 = load_afrag(sh1, 1);
      v16h a20 = load_afrag(sh2, 0);
      v16h a21 = load_afrag(sh2, 1);
#pragma unroll
      for (int nt = 0; nt < 2; ++nt) {
        v8f c = 0.0f;
        c = wmma_f16(a10, Bih2[nt][0], c);
        c = wmma_f16(a11, Bih2[nt][1], c);
        c = wmma_f16(a20, Bhh2[nt][0], c);
        c = wmma_f16(a21, Bhh2[nt][1], c);
        store_ctile(sg, c, ncolbase + nt * 16);
      }
    }
    __syncthreads();

    // ================= layer 2 elementwise + output dot =================
    {
      const float* gr = sg + er * SGS;
      float acc = 0.0f;
#pragma unroll
      for (int j = 0; j < 4; ++j) {
        const int c0 = ec + j;
        float ig = gr[c0]       + sb2[c0];
        float fg = gr[64 + c0]  + sb2[64 + c0];
        float gg = gr[128 + c0] + sb2[128 + c0];
        float og = gr[192 + c0] + sb2[192 + c0];
        float cn = fast_sigmoid(fg) * c2[j] + fast_sigmoid(ig) * fast_tanh(gg);
        c2[j] = cn;
        float hn = fast_sigmoid(og) * fast_tanh(cn);
        sh2[er * HS + c0] = (_Float16)hn;
        acc += hn * swout[c0];
      }
      // reduce the 64-wide dot across the 16 lanes that share row `er`
#pragma unroll
      for (int off = 8; off > 0; off >>= 1) acc += __shfl_down(acc, off, 16);
      if ((tid & 15) == 0) out[(size_t)(bbase + er) * T + t] = acc + bo;
    }
    __syncthreads();
  }
}

// ---- launch --------------------------------------------------------------
extern "C" void kernel_launch(void* const* d_in, const int* in_sizes, int n_in,
                              void* d_out, int out_size, void* d_ws, size_t ws_size,
                              hipStream_t stream) {
  const float* input = (const float*)d_in[0];
  const float* Wih1  = (const float*)d_in[1];
  const float* Whh1  = (const float*)d_in[2];
  const float* b1    = (const float*)d_in[3];
  const float* Wih2  = (const float*)d_in[4];
  const float* Whh2  = (const float*)d_in[5];
  const float* b2    = (const float*)d_in[6];
  const float* Wout  = (const float*)d_in[7];
  const float* bout  = (const float*)d_in[8];
  float* out = (float*)d_out;

  const int T = 1024;                    // sequence length (reference)
  const int B = in_sizes[0] / T;         // 2048
  const int grid = B / BT;               // 128 workgroups of 16 batch rows

  lstm2_wmma_kernel<<<dim3(grid), dim3(NW * 32), 0, stream>>>(
      input, Wih1, Whh1, b1, Wih2, Whh2, b2, Wout, bout, out, T);
}
